// WSI_GAT_40235253629415
// MI455X (gfx1250) — matrix operand order repaired and checked by reading, not compiled, not run on hardware
//
#include <hip/hip_runtime.h>
#include <hip/hip_bf16.h>

typedef __attribute__((ext_vector_type(16))) _Float16 v16h;
typedef __attribute__((ext_vector_type(8)))  _Float16 v8h;
typedef __attribute__((ext_vector_type(8)))  float    v8f;

// ---------------- problem constants ----------------
constexpr int N_NODES = 20000;
constexpr int E_EDGES = 320000;
constexpr int ETOT    = E_EDGES + N_NODES;   // with self loops
constexpr int IN_DIM  = 768;
constexpr int D       = 1024;
constexpr int HEADS   = 16;
constexpr int HID     = 64;
constexpr int NGRAPH  = 4;
constexpr float NEG_SLOPE = 0.2f;
constexpr float EPS = 1e-5f;

constexpr int ROWS_PER_BLOCK = 128;
constexpr int ROW_CHUNKS = (N_NODES + ROWS_PER_BLOCK - 1) / ROWS_PER_BLOCK;

// ---------------- small utility kernels ----------------
__global__ void fill_f32_kernel(float* p, float v, int n) {
  int i = blockIdx.x * blockDim.x + threadIdx.x;
  if (i < n) p[i] = v;
}
__global__ void fill_i32_kernel(int* p, int v, int n) {
  int i = blockIdx.x * blockDim.x + threadIdx.x;
  if (i < n) p[i] = v;
}
__global__ void f32_to_f16_kernel(const float* __restrict__ src, _Float16* __restrict__ dst, int n) {
  int i = blockIdx.x * blockDim.x + threadIdx.x;
  if (i < n) dst[i] = (_Float16)src[i];
}
// W [K][Nc] f32 -> Wt [Nc][K] f16, 32x32 tiles through LDS (coalesced both sides)
__global__ __launch_bounds__(256)
void transpose_f32_to_f16_kernel(const float* __restrict__ W, _Float16* __restrict__ Wt,
                                 int K, int Nc) {
  __shared__ float tile[32][33];
  const int tx = threadIdx.x & 31;
  const int ty = threadIdx.x >> 5;           // 0..7
  const int k0 = blockIdx.x * 32;
  const int n0 = blockIdx.y * 32;
  #pragma unroll
  for (int j = 0; j < 32; j += 8)
    tile[ty + j][tx] = W[(size_t)(k0 + ty + j) * Nc + (n0 + tx)];
  __syncthreads();
  #pragma unroll
  for (int j = 0; j < 32; j += 8)
    Wt[(size_t)(n0 + ty + j) * K + (k0 + tx)] = (_Float16)tile[tx][ty + j];
}
// edge_index int64 [2,E]; append self loops
__global__ void build_edges_kernel(const long long* __restrict__ ei, int* __restrict__ src,
                                   int* __restrict__ dst) {
  int i = blockIdx.x * blockDim.x + threadIdx.x;
  if (i >= ETOT) return;
  if (i < E_EDGES) {
    src[i] = (int)ei[i];
    dst[i] = (int)ei[E_EDGES + i];
  } else {
    src[i] = i - E_EDGES;
    dst[i] = i - E_EDGES;
  }
}

// ---------------- WMMA f16 GEMM: C[M,Nc] = A[M,K] x Wt[Nc,K]^T ----------------
constexpr int BM = 128, BN = 128, BK = 64, PAD = 8;

__global__ __launch_bounds__(256)
void gemm_f16_wmma_kernel(const _Float16* __restrict__ A, const _Float16* __restrict__ Bt,
                          float* __restrict__ C, int M, int K, int Nc) {
  __shared__ _Float16 As [BM][BK + PAD];   // row-major  [m][k]
  __shared__ _Float16 Bst[BN][BK + PAD];   // transposed [n][k]

  const int tid   = threadIdx.x;
  const int lane  = tid & 31;
  const int wid   = tid >> 5;   // 8 waves
  const int wm    = wid & 3;    // 4 wave rows  -> 32 M rows each
  const int wn    = wid >> 2;   // 2 wave cols  -> 64 N cols each
  const int l16   = lane & 15;
  const int lhalf = lane >> 4;  // 0 / 1

  const int blockRow = blockIdx.y * BM;
  const int blockCol = blockIdx.x * BN;

  const v8f vzero = {0.f, 0.f, 0.f, 0.f, 0.f, 0.f, 0.f, 0.f};
  const v8h hzero = {(_Float16)0.f, (_Float16)0.f, (_Float16)0.f, (_Float16)0.f,
                     (_Float16)0.f, (_Float16)0.f, (_Float16)0.f, (_Float16)0.f};
  v8f acc[2][4];
  #pragma unroll
  for (int mt = 0; mt < 2; ++mt)
    #pragma unroll
    for (int nt = 0; nt < 4; ++nt) acc[mt][nt] = vzero;

  for (int k0 = 0; k0 < K; k0 += BK) {
    // stage A tile (BM x BK) as 8-elem (16B) vectors: 4 per thread
    #pragma unroll
    for (int i = 0; i < (BM * BK) / (256 * 8); ++i) {
      int u = i * 256 + tid;              // 8-elem unit index
      int r = u >> 3;                     // / (BK/8)
      int c = (u & 7) * 8;
      int gr = blockRow + r;
      *(v8h*)&As[r][c] =
          (gr < M) ? *(const v8h*)&A[(size_t)gr * K + (k0 + c)] : hzero;
    }
    // stage B tile (BN x BK) from pre-transposed Wt[n][k]: 4 per thread, all 16B
    #pragma unroll
    for (int i = 0; i < (BN * BK) / (256 * 8); ++i) {
      int u = i * 256 + tid;
      int r = u >> 3;
      int c = (u & 7) * 8;
      *(v8h*)&Bst[r][c] = *(const v8h*)&Bt[(size_t)(blockCol + r) * K + (k0 + c)];
    }
    __syncthreads();

    #pragma unroll
    for (int ks = 0; ks < BK / 32; ++ks) {
      // A fragments: lanes 0-15 -> K kb..kb+7 / kb+16..kb+23 (kb = ks*32 [+8 for hi half])
      v16h afrag[2];
      #pragma unroll
      for (int mt = 0; mt < 2; ++mt) {
        const int arow = wm * 32 + mt * 16 + l16;
        const int kb = ks * 32 + lhalf * 8;
        v8h lo = *(const v8h*)&As[arow][kb];
        v8h hi = *(const v8h*)&As[arow][kb + 16];
        afrag[mt] = __builtin_shufflevector(lo, hi, 0, 1, 2, 3, 4, 5, 6, 7,
                                            8, 9, 10, 11, 12, 13, 14, 15);
      }
      // B fragments: lane holds column n = lane%16, k = kb..kb+7 / kb+16..kb+23 (kb = ks*32 [+8 for hi half])
      v16h bfrag[4];
      #pragma unroll
      for (int nt = 0; nt < 4; ++nt) {
        const int bcol = wn * 64 + nt * 16 + l16;
        const int kb = ks * 32 + lhalf * 8;
        v8h lo = *(const v8h*)&Bst[bcol][kb];
        v8h hi = *(const v8h*)&Bst[bcol][kb + 16];
        bfrag[nt] = __builtin_shufflevector(lo, hi, 0, 1, 2, 3, 4, 5, 6, 7,
                                            8, 9, 10, 11, 12, 13, 14, 15);
      }
      #pragma unroll
      for (int mt = 0; mt < 2; ++mt)
        #pragma unroll
        for (int nt = 0; nt < 4; ++nt)
          acc[mt][nt] = __builtin_amdgcn_wmma_f32_16x16x32_f16(
              false, afrag[mt], false, bfrag[nt], (short)0, acc[mt][nt], false, false);
    }
    __syncthreads();
  }

  // C layout: lanes 0-15 -> M = v, lanes 16-31 -> M = 8+v; N = lane%16
  #pragma unroll
  for (int mt = 0; mt < 2; ++mt) {
    const int rbase = blockRow + wm * 32 + mt * 16 + lhalf * 8;
    #pragma unroll
    for (int nt = 0; nt < 4; ++nt) {
      const int col = blockCol + wn * 64 + nt * 16 + l16;
      #pragma unroll
      for (int v = 0; v < 8; ++v) {
        int r = rbase + v;
        if (r < M) C[(size_t)r * Nc + col] = acc[mt][nt][v];
      }
    }
  }
}

// ---------------- per-node attention scores ----------------
__global__ __launch_bounds__(256)
void scores_kernel(const float* __restrict__ h, const float* __restrict__ asrc,
                   const float* __restrict__ adst, float* __restrict__ es,
                   float* __restrict__ ed, int H, int C) {
  const int n = blockIdx.x;
  const int tid = threadIdx.x;
  __shared__ float r0[256];
  __shared__ float r1[256];
  const float* hn = h + (size_t)n * (H * C);
  for (int hh = 0; hh < H; ++hh) {
    float s0 = 0.f, s1 = 0.f;
    for (int c = tid; c < C; c += 256) {
      float v = hn[hh * C + c];
      s0 += v * asrc[hh * C + c];
      s1 += v * adst[hh * C + c];
    }
    r0[tid] = s0; r1[tid] = s1;
    __syncthreads();
    for (int off = 128; off > 0; off >>= 1) {
      if (tid < off) { r0[tid] += r0[tid + off]; r1[tid] += r1[tid + off]; }
      __syncthreads();
    }
    if (tid == 0) { es[n * H + hh] = r0[0]; ed[n * H + hh] = r1[0]; }
    __syncthreads();
  }
}

// ---------------- edge softmax (ordered-int float max trick) ----------------
__device__ __forceinline__ int enc_f(float f) {
  int i = __float_as_int(f);
  return (i >= 0) ? i : (i ^ 0x7fffffff);
}
__device__ __forceinline__ float dec_f(int i) {
  return __int_as_float((i >= 0) ? i : (i ^ 0x7fffffff));
}
__device__ __forceinline__ float lrelu(float v) { return (v > 0.f) ? v : NEG_SLOPE * v; }

__global__ void edge_max_kernel(const int* __restrict__ src, const int* __restrict__ dst,
                                const float* __restrict__ es, const float* __restrict__ ed,
                                int* __restrict__ mmax, int H) {
  int t = blockIdx.x * blockDim.x + threadIdx.x;
  if (t >= ETOT * H) return;
  int e = t / H, hh = t - e * H;
  int s = src[e], d = dst[e];
  float v = lrelu(es[s * H + hh] + ed[d * H + hh]);
  atomicMax(&mmax[d * H + hh], enc_f(v));
}

__global__ void edge_denom_kernel(const int* __restrict__ src, const int* __restrict__ dst,
                                  const float* __restrict__ es, const float* __restrict__ ed,
                                  const int* __restrict__ mmax, float* __restrict__ denom, int H) {
  int t = blockIdx.x * blockDim.x + threadIdx.x;
  if (t >= ETOT * H) return;
  int e = t / H, hh = t - e * H;
  int s = src[e], d = dst[e];
  float v = lrelu(es[s * H + hh] + ed[d * H + hh]);
  float p = __expf(v - dec_f(mmax[d * H + hh]));
  atomicAdd(&denom[d * H + hh], p);
}

// one block per edge, 256 threads x 4 channels = D; unnormalized p-weighted scatter
__global__ __launch_bounds__(256)
void edge_agg_kernel(const int* __restrict__ src, const int* __restrict__ dst,
                     const float* __restrict__ es, const float* __restrict__ ed,
                     const int* __restrict__ mmax, const float* __restrict__ hlin,
                     float* __restrict__ agg, int H, int C) {
  const int e = blockIdx.x;
  const int s = src[e], d = dst[e];
  const int c0 = threadIdx.x * 4;          // D == 1024
  const int hh = c0 / C;
  float v = lrelu(es[s * H + hh] + ed[d * H + hh]);
  float p = __expf(v - dec_f(mmax[d * H + hh]));
  const float4 hv = *(const float4*)(hlin + (size_t)s * D + c0);
  float* ap = agg + (size_t)d * D + c0;
  atomicAdd(ap + 0, p * hv.x);
  atomicAdd(ap + 1, p * hv.y);
  atomicAdd(ap + 2, p * hv.z);
  atomicAdd(ap + 3, p * hv.w);
}

// ---------------- normalize by denom + bias, accumulate BN stats ----------------
__global__ __launch_bounds__(256)
void norm_stats_kernel(float* __restrict__ agg, const float* __restrict__ denom,
                       const float* __restrict__ bias, float* __restrict__ bnsum,
                       float* __restrict__ bnsq, int H) {
  const int c = blockIdx.x * 256 + threadIdx.x;          // c < D
  const int C = D / H;
  const int hh = c / C;
  const int r0 = blockIdx.y * ROWS_PER_BLOCK;
  const int r1 = min(r0 + ROWS_PER_BLOCK, N_NODES);
  float s = 0.f, sq = 0.f;
  for (int r = r0; r < r1; ++r) {
    float v = agg[(size_t)r * D + c] / denom[r * H + hh] + bias[c];
    agg[(size_t)r * D + c] = v;
    s += v; sq += v * v;
  }
  atomicAdd(&bnsum[c], s);
  atomicAdd(&bnsq[c], sq);
}

// ---------------- BN + ReLU (+ residual) ----------------
__global__ __launch_bounds__(256)
void bn_apply_kernel(const float* __restrict__ agg, const float* __restrict__ bnsum,
                     const float* __restrict__ bnsq, const float* __restrict__ gamma,
                     const float* __restrict__ beta, const float* __restrict__ residual,
                     float* __restrict__ out) {
  const int c = blockIdx.x * 256 + threadIdx.x;
  const float mean = bnsum[c] / (float)N_NODES;
  const float var  = bnsq[c] / (float)N_NODES - mean * mean;
  const float inv  = rsqrtf(var + EPS) * gamma[c];
  const float bet  = beta[c];
  const int r0 = blockIdx.y * ROWS_PER_BLOCK;
  const int r1 = min(r0 + ROWS_PER_BLOCK, N_NODES);
  for (int r = r0; r < r1; ++r) {
    size_t idx = (size_t)r * D + c;
    float v = (agg[idx] - mean) * inv + bet;
    v = fmaxf(v, 0.f);
    if (residual) v += residual[idx];
    out[idx] = v;
  }
}

// ---------------- graph mean pool ----------------
__global__ void count_kernel(const long long* __restrict__ batch, float* __restrict__ counts) {
  int i = blockIdx.x * blockDim.x + threadIdx.x;
  if (i < N_NODES) atomicAdd(&counts[(int)batch[i]], 1.f);
}
__global__ __launch_bounds__(256)
void pool_sum_kernel(const float* __restrict__ h, const long long* __restrict__ batch,
                     float* __restrict__ out) {
  const int c = blockIdx.x * 256 + threadIdx.x;
  const int r0 = blockIdx.y * ROWS_PER_BLOCK;
  const int r1 = min(r0 + ROWS_PER_BLOCK, N_NODES);
  float acc[NGRAPH] = {0.f, 0.f, 0.f, 0.f};
  for (int r = r0; r < r1; ++r)
    acc[(int)batch[r]] += h[(size_t)r * D + c];
  #pragma unroll
  for (int g = 0; g < NGRAPH; ++g)
    atomicAdd(&out[(size_t)g * D + c], acc[g]);
}
__global__ void pool_div_kernel(float* __restrict__ out, const float* __restrict__ counts) {
  int i = blockIdx.x * blockDim.x + threadIdx.x;
  if (i < NGRAPH * D) out[i] /= counts[i / D];
}

// ---------------- host driver ----------------
extern "C" void kernel_launch(void* const* d_in, const int* in_sizes, int n_in,
                              void* d_out, int out_size, void* d_ws, size_t ws_size,
                              hipStream_t stream) {
  const float*      x        = (const float*)d_in[0];
  const long long*  ei       = (const long long*)d_in[1];
  const long long*  batch    = (const long long*)d_in[2];
  const float*      W0       = (const float*)d_in[3];
  const float*      att_src0 = (const float*)d_in[4];
  const float*      att_dst0 = (const float*)d_in[5];
  const float*      b0       = (const float*)d_in[6];
  const float*      Ws       = (const float*)d_in[7];
  const float*      att_srcs = (const float*)d_in[8];
  const float*      att_dsts = (const float*)d_in[9];
  const float*      bs       = (const float*)d_in[10];
  const float*      gammas   = (const float*)d_in[11];
  const float*      betas    = (const float*)d_in[12];
  float*            out      = (float*)d_out;

  // workspace bump allocator (256B aligned)
  char* wp = (char*)d_ws;
  auto take = [&](size_t bytes) -> void* {
    void* r = (void*)wp;
    wp += (bytes + 255) & ~(size_t)255;
    return r;
  };
  int*      src32  = (int*)take((size_t)ETOT * 4);
  int*      dst32  = (int*)take((size_t)ETOT * 4);
  _Float16* actf16 = (_Float16*)take((size_t)N_NODES * D * 2);
  _Float16* wtf16  = (_Float16*)take((size_t)D * D * 2);    // transposed weights [Nc][K]
  float*    hlin   = (float*)take((size_t)N_NODES * D * 4); // GEMM output h
  float*    agg    = (float*)take((size_t)N_NODES * D * 4); // aggregation accumulator
  float*    hcur   = (float*)take((size_t)N_NODES * D * 4); // layer input / residual
  float*    es     = (float*)take((size_t)N_NODES * HEADS * 4);
  float*    ed     = (float*)take((size_t)N_NODES * HEADS * 4);
  int*      mmax   = (int*)take((size_t)N_NODES * HEADS * 4);
  float*    denom  = (float*)take((size_t)N_NODES * HEADS * 4);
  float*    bnsum  = (float*)take((size_t)D * 4);
  float*    bnsq   = (float*)take((size_t)D * 4);
  float*    counts = (float*)take((size_t)NGRAPH * 4);

  auto cdiv = [](int a, int b) { return (a + b - 1) / b; };

  build_edges_kernel<<<cdiv(ETOT, 256), 256, 0, stream>>>(ei, src32, dst32);

  for (int layer = 0; layer < 4; ++layer) {
    const int H = (layer == 0) ? HEADS : 1;
    const int C = D / H;
    const int K = (layer == 0) ? IN_DIM : D;
    const float* Af32  = (layer == 0) ? x : hcur;
    const float* Wf32  = (layer == 0) ? W0 : (Ws + (size_t)(layer - 1) * D * D);
    const float* asrc  = (layer == 0) ? att_src0 : (att_srcs + (size_t)(layer - 1) * D);
    const float* adst  = (layer == 0) ? att_dst0 : (att_dsts + (size_t)(layer - 1) * D);
    const float* bias  = (layer == 0) ? b0 : (bs + (size_t)(layer - 1) * D);
    const float* gamma = gammas + (size_t)layer * D;
    const float* beta  = betas + (size_t)layer * D;
    const float* resid = (layer == 0) ? nullptr : hcur;

    // activations f32 -> f16 (row-major), weights f32 -> f16 transposed [Nc][K]
    f32_to_f16_kernel<<<cdiv(N_NODES * K, 256), 256, 0, stream>>>(Af32, actf16, N_NODES * K);
    dim3 tgrid(K / 32, D / 32);
    transpose_f32_to_f16_kernel<<<tgrid, 256, 0, stream>>>(Wf32, wtf16, K, D);

    // GEMM: hlin = A x W  (f16 in, f32 accumulate via v_wmma_f32_16x16x32_f16)
    dim3 ggrid(D / BN, cdiv(N_NODES, BM));
    gemm_f16_wmma_kernel<<<ggrid, 256, 0, stream>>>(actf16, wtf16, hlin, N_NODES, K, D);

    // attention scores
    scores_kernel<<<N_NODES, 256, 0, stream>>>(hlin, asrc, adst, es, ed, H, C);

    // edge softmax
    fill_i32_kernel<<<cdiv(N_NODES * H, 256), 256, 0, stream>>>(mmax, (int)0x80000000, N_NODES * H);
    fill_f32_kernel<<<cdiv(N_NODES * H, 256), 256, 0, stream>>>(denom, 0.f, N_NODES * H);
    fill_f32_kernel<<<cdiv(N_NODES * D, 256), 256, 0, stream>>>(agg, 0.f, N_NODES * D);
    edge_max_kernel<<<cdiv(ETOT * H, 256), 256, 0, stream>>>(src32, dst32, es, ed, mmax, H);
    edge_denom_kernel<<<cdiv(ETOT * H, 256), 256, 0, stream>>>(src32, dst32, es, ed, mmax, denom, H);
    edge_agg_kernel<<<ETOT, 256, 0, stream>>>(src32, dst32, es, ed, mmax, hlin, agg, H, C);

    // normalize + bias + BN stats, then BN + ReLU (+ residual) -> hcur
    fill_f32_kernel<<<cdiv(D, 256), 256, 0, stream>>>(bnsum, 0.f, D);
    fill_f32_kernel<<<cdiv(D, 256), 256, 0, stream>>>(bnsq, 0.f, D);
    dim3 rgrid(D / 256, ROW_CHUNKS);
    norm_stats_kernel<<<rgrid, 256, 0, stream>>>(agg, denom, bias, bnsum, bnsq, H);
    bn_apply_kernel<<<rgrid, 256, 0, stream>>>(agg, bnsum, bnsq, gamma, beta, resid, hcur);
  }

  // graph mean pool -> out[4,1024]
  fill_f32_kernel<<<cdiv(NGRAPH * D, 256), 256, 0, stream>>>(out, 0.f, NGRAPH * D);
  fill_f32_kernel<<<1, 256, 0, stream>>>(counts, 0.f, NGRAPH);
  count_kernel<<<cdiv(N_NODES, 256), 256, 0, stream>>>(batch, counts);
  dim3 pgrid(D / 256, ROW_CHUNKS);
  pool_sum_kernel<<<pgrid, 256, 0, stream>>>(hcur, batch, out);
  pool_div_kernel<<<cdiv(NGRAPH * D, 256), 256, 0, stream>>>(out, counts);

  (void)in_sizes; (void)n_in; (void)out_size; (void)ws_size;
}
